// TF_SAM_63780264346004
// MI455X (gfx1250) — compile-verified, matrix operand order checked by reading, not run on hardware
//
#include <hip/hip_runtime.h>
#include <hip/hip_bf16.h>

typedef __attribute__((ext_vector_type(16))) _Float16 v16h;
typedef __attribute__((ext_vector_type(8)))  _Float16 v8h;
typedef __attribute__((ext_vector_type(8)))  float    v8f;
typedef __attribute__((ext_vector_type(4)))  unsigned int v4u;
typedef __attribute__((ext_vector_type(4)))  int v4si;
typedef __attribute__((ext_vector_type(8)))  int v8si;

#define BN 8192
#define CDIM 256
#define EPSN 1e-12f
#define WBLEND 0.8f
#define KT 32
#define NITER (BN / KT)

#if __has_builtin(__builtin_amdgcn_tensor_load_to_lds)
#define HAVE_TDM 1
#else
#define HAVE_TDM 0
#endif

// ---------------------------------------------------------------------------
// 16-bit WMMA A/B fragment loader (wave32 layout, ISA 7.12.2): lane holds row
// m = lane&15; K-chunks [kh,kh+8) and [16+kh,16+kh+8), kh = (lane>>4)*8.
// Works for global or LDS pointers (addrspace inferred; 2x 16B loads).
// ---------------------------------------------------------------------------
__device__ __forceinline__ v16h frag_ld(const _Float16* p, int ld) {
    int lane = threadIdx.x & 31;
    int m    = lane & 15;
    int kh   = (lane >> 4) << 3;
    const _Float16* r = p + (size_t)m * ld + kh;
    v8h lo  = *(const v8h*)(r);
    v8h hi8 = *(const v8h*)(r + 16);
    v16h out;
#pragma unroll
    for (int j = 0; j < 8; ++j) { out[j] = lo[j]; out[8 + j] = hi8[j]; }
    return out;
}

#if HAVE_TDM
// ---------------------------------------------------------------------------
// Tensor Data Mover: 2-D tile load Global -> LDS (D# per ISA ch.8).
// data_size = 2 bytes; dims/strides in element units; global_addr in bytes.
// ---------------------------------------------------------------------------
__device__ __forceinline__ void tdm_load_2d(
    unsigned lds_off, const _Float16* gptr,
    unsigned tensor_d0, unsigned tensor_d1,
    unsigned tile_d0, unsigned tile_d1, unsigned stride0)
{
    unsigned long long ga = (unsigned long long)(uintptr_t)gptr;
    v4u g0;
    g0[0] = 1u;                                   // count=1, user mode
    g0[1] = lds_off;                              // lds_addr
    g0[2] = (unsigned)(ga & 0xffffffffu);         // global_addr[31:0]
    g0[3] = (unsigned)(ga >> 32) | (2u << 30);    // global_addr[56:32] | type=2

    unsigned w0 = 0x00010000u;                    // data_size=1 (2 bytes)
    unsigned w1 = (tensor_d0 & 0xffffu) << 16;    // tensor_dim0[15:0]
    unsigned w2 = (tensor_d0 >> 16) | ((tensor_d1 & 0xffffu) << 16);
    unsigned w3 = (tensor_d1 >> 16) | (tile_d0 << 16);
    unsigned w4 = tile_d1;                        // tile_dim1 | tile_dim2=0
    unsigned w5 = stride0;                        // tensor_dim0_stride[31:0]
    v8si g1 = { (int)w0, (int)w1, (int)w2, (int)w3,
                (int)w4, (int)w5, 0, 0 };
    v4si z4 = { 0, 0, 0, 0 };
#if defined(__clang_major__) && (__clang_major__ >= 23)
    v8si z8 = { 0, 0, 0, 0, 0, 0, 0, 0 };
    __builtin_amdgcn_tensor_load_to_lds(g0, g1, z4, z4, z8, 0);
#else
    __builtin_amdgcn_tensor_load_to_lds(g0, g1, z4, z4, 0);
#endif
}
#endif

// ---------------------------------------------------------------------------
// Kernel 1: row L2-normalize.  One wave per row (8 rows / 256-thread block).
// Emits fnorm (f32 blend source), f16 (row-major Q/K), fT (channel-major V).
// ---------------------------------------------------------------------------
__global__ __launch_bounds__(256) void tf_sam_norm(
    const float* __restrict__ feats,
    float* __restrict__ fnorm,
    _Float16* __restrict__ f16,
    _Float16* __restrict__ fT)
{
    int wave = threadIdx.x >> 5;
    int lane = threadIdx.x & 31;
    int row  = blockIdx.x * 8 + wave;

    const float* src = feats + (size_t)row * CDIM + lane * 8;
    v8f v = *(const v8f*)src;
    float ss = 0.f;
#pragma unroll
    for (int j = 0; j < 8; ++j) ss += v[j] * v[j];
#pragma unroll
    for (int off = 16; off >= 1; off >>= 1) ss += __shfl_xor(ss, off, 32);
    float inv = 1.0f / fmaxf(sqrtf(ss), EPSN);

    v8f o; v8h oh;
#pragma unroll
    for (int j = 0; j < 8; ++j) {
        float x = v[j] * inv;
        o[j]  = x;
        oh[j] = (_Float16)x;
        fT[(size_t)(lane * 8 + j) * BN + row] = (_Float16)x;
    }
    *(v8f*)(fnorm + (size_t)row * CDIM + lane * 8) = o;
    *(v8h*)(f16   + (size_t)row * CDIM + lane * 8) = oh;
}

// ---------------------------------------------------------------------------
// Kernel 2: fused flash attention + blend + renormalize.
// 8 waves/block, 128 query rows/block, 64 blocks.  K/V tiles (32 keys) staged
// in LDS via TDM (double-buffered) and shared by all 8 waves -> 8x less L2
// traffic.  Per iter/wave: 16 WMMA (S=QK^T) + online softmax + 16 WMMA (PV).
// LDS halves layout: K0[8192] K1[8192] V0[8192] V1[8192] P[8][512]
// ---------------------------------------------------------------------------
__global__ __launch_bounds__(256) void tf_sam_attn(
    const float* __restrict__ fnorm,
    const _Float16* __restrict__ f16,
    const _Float16* __restrict__ fT,
    float* __restrict__ out)
{
    extern __shared__ _Float16 smem[];

    int wave  = threadIdx.x >> 5;
    int lane  = threadIdx.x & 31;
    int n     = lane & 15;
    int hi    = lane >> 4;
    int qbase = blockIdx.x * 128 + wave * 16;

    _Float16* pl = smem + 32768 + wave * (16 * KT);

    // Cache Q A-fragments (8 k-steps of the 256-wide channel dim).
    v16h qf[8];
#pragma unroll
    for (int kk = 0; kk < 8; ++kk)
        qf[kk] = frag_ld(f16 + (size_t)qbase * CDIM + kk * 32, CDIM);

    v8f acc[16];
#pragma unroll
    for (int t = 0; t < 16; ++t) acc[t] = (v8f){};
    float mrow[8], lrow[8];
#pragma unroll
    for (int r = 0; r < 8; ++r) { mrow[r] = -3.0e38f; lrow[r] = 0.f; }

#if HAVE_TDM
    unsigned smem_off = (unsigned)(uintptr_t)smem;   // LDS byte offset of base
    if (wave == 0) {   // prologue: fill buffer 0
        tdm_load_2d(smem_off,               f16, CDIM, BN, CDIM, KT, CDIM);
        tdm_load_2d(smem_off + 16384u * 2u, fT,  BN, CDIM, KT, CDIM, BN);
    }
#endif

    for (int i = 0; i < NITER; ++i) {
        unsigned cur = (unsigned)(i & 1);
        const _Float16* Kl = smem + cur * 8192u;
        const _Float16* Vl = smem + 16384u + cur * 8192u;

#if HAVE_TDM
        if (wave == 0) {
            if (i + 1 < NITER) {
                unsigned nxt = cur ^ 1u;
                int kb2 = (i + 1) * KT;
                tdm_load_2d(smem_off + nxt * 8192u * 2u,
                            f16 + (size_t)kb2 * CDIM, CDIM, BN, CDIM, KT, CDIM);
                tdm_load_2d(smem_off + (16384u + nxt * 8192u) * 2u,
                            fT + kb2, BN, CDIM, KT, CDIM, BN);
                __builtin_amdgcn_s_wait_tensorcnt((short)2);  // current tile done
            } else {
                __builtin_amdgcn_s_wait_tensorcnt((short)0);
            }
        }
        __syncthreads();                       // publish buf[cur] to all waves
#else
        {   // fallback staging: 256 threads x (16B K + 16B V) x 4
            int kb = i * KT;
            int t  = threadIdx.x;
            _Float16* Kw = smem + cur * 8192u;
            _Float16* Vw = smem + 16384u + cur * 8192u;
#pragma unroll
            for (int j = 0; j < 4; ++j) {
                int idx = t + 256 * j;                       // 0..1023 (16B chunks)
                *(v8h*)(Kw + idx * 8) =
                    *(const v8h*)(f16 + (size_t)kb * CDIM + idx * 8);
                int row = idx >> 2, c16 = idx & 3;
                *(v8h*)(Vw + row * KT + c16 * 8) =
                    *(const v8h*)(fT + (size_t)row * BN + kb + c16 * 8);
            }
        }
        __syncthreads();
#endif

        // ---- S = Q K^T over two 16-key tiles, channel dim in 8 k-steps ----
        v8f s0 = (v8f){}, s1 = (v8f){};
#pragma unroll
        for (int kk = 0; kk < 8; ++kk) {
            v16h b0 = frag_ld(Kl + kk * 32, CDIM);
            v16h b1 = frag_ld(Kl + 16 * CDIM + kk * 32, CDIM);
            s0 = __builtin_amdgcn_wmma_f32_16x16x32_f16(false, qf[kk], false, b0,
                                                        (short)0, s0, false, false);
            s1 = __builtin_amdgcn_wmma_f32_16x16x32_f16(false, qf[kk], false, b1,
                                                        (short)0, s1, false, false);
        }

        // ---- online softmax (TEMPERATURE == 1.0) ----
#pragma unroll
        for (int r = 0; r < 8; ++r) {
            float mx = fmaxf(s0[r], s1[r]);
            mx = fmaxf(mx, __shfl_xor(mx, 1, 32));
            mx = fmaxf(mx, __shfl_xor(mx, 2, 32));
            mx = fmaxf(mx, __shfl_xor(mx, 4, 32));
            mx = fmaxf(mx, __shfl_xor(mx, 8, 32));
            float mnew = fmaxf(mrow[r], mx);
            float corr = __expf(mrow[r] - mnew);
            mrow[r] = mnew;

            float p0 = __expf(s0[r] - mnew);
            float p1 = __expf(s1[r] - mnew);
            float rs = p0 + p1;
            rs += __shfl_xor(rs, 1, 32);
            rs += __shfl_xor(rs, 2, 32);
            rs += __shfl_xor(rs, 4, 32);
            rs += __shfl_xor(rs, 8, 32);
            lrow[r] = lrow[r] * corr + rs;

#pragma unroll
            for (int t = 0; t < 16; ++t) acc[t][r] *= corr;

            int prow = r + 8 * hi;             // C-layout -> row-major P in LDS
            pl[prow * KT + n]      = (_Float16)p0;
            pl[prow * KT + 16 + n] = (_Float16)p1;
        }

        // same-wave LDS RAW on P buffer: DS ops are in-order, wait for stores
        asm volatile("s_wait_dscnt 0" ::: "memory");

        // ---- acc += P V over 16 channel tiles ----
        v16h ap = frag_ld(pl, KT);
#pragma unroll
        for (int t = 0; t < 16; ++t) {
            v16h bv = frag_ld(Vl + (t * 16) * KT, KT);
            acc[t] = __builtin_amdgcn_wmma_f32_16x16x32_f16(false, ap, false, bv,
                                                            (short)0, acc[t], false, false);
        }

        __syncthreads();   // all waves done with buf[cur] before it is refilled
    }

    // ---- epilogue: divide by denom, blend, renormalize rows, store ----
#pragma unroll
    for (int r = 0; r < 8; ++r) {
        int row = qbase + r + 8 * hi;
        float linv = 1.0f / lrow[r];
        float ssq = 0.f;
#pragma unroll
        for (int t = 0; t < 16; ++t) {
            int c = t * 16 + n;
            float b = WBLEND * fnorm[(size_t)row * CDIM + c]
                    + (1.0f - WBLEND) * acc[t][r] * linv;
            acc[t][r] = b;
            ssq += b * b;
        }
        ssq += __shfl_xor(ssq, 1, 32);
        ssq += __shfl_xor(ssq, 2, 32);
        ssq += __shfl_xor(ssq, 4, 32);
        ssq += __shfl_xor(ssq, 8, 32);
        float rn = 1.0f / fmaxf(sqrtf(ssq), EPSN);
#pragma unroll
        for (int t = 0; t < 16; ++t)
            out[(size_t)row * CDIM + t * 16 + n] = acc[t][r] * rn;
    }
}

// ---------------------------------------------------------------------------
extern "C" void kernel_launch(void* const* d_in, const int* in_sizes, int n_in,
                              void* d_out, int out_size, void* d_ws, size_t ws_size,
                              hipStream_t stream) {
    const float* feats = (const float*)d_in[0];
    float* out = (float*)d_out;

    // workspace: fnorm (8 MB f32) | f16 row-major (4 MB) | fT channel-major (4 MB)
    float*    fnorm = (float*)d_ws;
    _Float16* f16   = (_Float16*)(fnorm + (size_t)BN * CDIM);
    _Float16* fT    = f16 + (size_t)BN * CDIM;

    tf_sam_norm<<<BN / 8, 256, 0, stream>>>(feats, fnorm, f16, fT);

    // dynamic LDS: 2x16KB K + 2x16KB V + 8x1KB P = 72 KB (CDNA5 WGP: 320 KB)
    size_t lds_bytes = (size_t)(2 * KT * CDIM + 2 * CDIM * KT + 8 * 16 * KT) * sizeof(_Float16);
    tf_sam_attn<<<BN / 128, 256, lds_bytes, stream>>>(fnorm, f16, fT, out);
}
